// DGCNN_61409442398409
// MI455X (gfx1250) — matrix operand order repaired and checked by reading, not compiled, not run on hardware
//
#include <hip/hip_runtime.h>

#define NP   1024
#define KNN  20
#define NB   16

typedef __attribute__((ext_vector_type(16))) __bf16       v16bf;
typedef __attribute__((ext_vector_type(8)))  float        v8f;
typedef __attribute__((ext_vector_type(8)))  unsigned int v8u;

// ---------------- helpers ----------------

static __device__ __forceinline__ unsigned short f2bf(float f) {
  unsigned u = __builtin_bit_cast(unsigned, f);
  unsigned r = u + 0x7FFFu + ((u >> 16) & 1u);   // round-to-nearest-even
  return (unsigned short)(r >> 16);
}
static __device__ __forceinline__ unsigned pk2(float a, float b) {
  return (unsigned)f2bf(a) | ((unsigned)f2bf(b) << 16);
}

// gfx1250 async copy: global -> LDS, 16 bytes per lane (ASYNCcnt tracked)
static __device__ __forceinline__ void async_b128(void* lds, const void* g) {
  unsigned loff = (unsigned)(unsigned long long)lds;   // low 32b of generic = LDS offset
  asm volatile("global_load_async_to_lds_b128 %0, %1, off"
               :: "v"(loff), "v"(g) : "memory");
}
// graded wait: ASYNCcnt completes in-order, so waiting down to the number of
// ops issued for the *next* chunk guarantees the *current* chunk has landed.
static __device__ __forceinline__ void wait_async(int n) {
  if (n == 0)      asm volatile("s_wait_asynccnt 0x0" ::: "memory");
  else if (n == 1) asm volatile("s_wait_asynccnt 0x1" ::: "memory");
  else             asm volatile("s_wait_asynccnt 0x2" ::: "memory");
}

// LDS tile row pitch: 40 ushorts = 80 B (16B-aligned rows for async B128)
#define PITCH 40

// A-matrix 16x32 bf16 fragment: lane L holds row m=L%16; K-half h=L/16.
static __device__ __forceinline__ v16bf frag_a(const unsigned int* p, int h) {
  v8u u; const int o = 4 * h;
  u[0] = p[o + 0]; u[1] = p[o + 1]; u[2] = p[o + 2]; u[3] = p[o + 3];
  u[4] = p[o + 8]; u[5] = p[o + 9]; u[6] = p[o + 10]; u[7] = p[o + 11];
  return __builtin_bit_cast(v16bf, u);
}
// B-matrix 32x16 bf16 fragment: lane L holds col n=L%16; K-half h=L/16 (same K order as A).
static __device__ __forceinline__ v16bf frag_b(const unsigned int* p, int h) {
  v8u u; const int o = 4 * h;
#pragma unroll
  for (int j = 0; j < 8; ++j) u[j] = p[o + (j < 4 ? j : j + 4)];
  return __builtin_bit_cast(v16bf, u);
}

// ---------------- squared norms ----------------

template <int C>
__global__ void norms_kernel(const float* __restrict__ xin, int ldx,
                             float* __restrict__ norms) {
  int t = blockIdx.x * 256 + threadIdx.x;          // B*N threads
  const float* p = xin + (size_t)t * ldx;
  float s = 0.f;
#pragma unroll
  for (int c = 0; c < C; ++c) s = fmaf(p[c], p[c], s);
  norms[t] = s;
}

// ---------------- kNN (top-20 smallest squared distance, self included) ----

template <int C>
__global__ void knn_kernel(const float* __restrict__ xin, int ldx,
                           const float* __restrict__ norms,
                           int* __restrict__ idxout) {
  int t = blockIdx.x * 128 + threadIdx.x;          // B*N threads
  int b = t >> 10;
  int n = t & (NP - 1);
  const float* xb = xin + ((size_t)b * NP) * ldx;
  float xn[C];
#pragma unroll
  for (int c = 0; c < C; ++c) xn[c] = xb[(size_t)n * ldx + c];
  float nn = norms[b * NP + n];

  float bd[KNN]; int bi[KNN];
#pragma unroll
  for (int j = 0; j < KNN; ++j) { bd[j] = 3.402823466e38f; bi[j] = 0; }

  for (int m = 0; m < NP; ++m) {
    const float* xm = xb + (size_t)m * ldx;
    float dot = 0.f;
#pragma unroll
    for (int c = 0; c < C; ++c) dot = fmaf(xn[c], xm[c], dot);
    float d = nn + norms[b * NP + m] - 2.f * dot;
    if (d < bd[KNN - 1]) {
      int id = m;
#pragma unroll
      for (int j = 0; j < KNN; ++j) {
        bool sw = d < bd[j];
        float td = bd[j]; int ti = bi[j];
        bd[j] = sw ? d : td;  bi[j] = sw ? id : ti;
        d     = sw ? td : d;  id    = sw ? ti : id;
      }
    }
  }
#pragma unroll
  for (int j = 0; j < KNN; ++j) idxout[(size_t)t * KNN + j] = bi[j];
}

// ---------------- weight fp32 -> bf16 (zero-padded to Kp) ------------------

__global__ void cvt_w(const float* __restrict__ W, int Cin, int Kp,
                      unsigned short* __restrict__ out, int total) {
  int t = blockIdx.x * 256 + threadIdx.x;
  if (t >= total) return;
  int o = t / Kp, c = t - o * Kp;
  out[t] = f2bf(c < Cin ? W[(size_t)o * Cin + c] : 0.f);
}

// ---------------- edge GEMM --------------------------------------------------
// out[row,o] = sum_c e[row,c]*W[o,c], e = [x[nbr]-x[n] (C) | x[n] (C)]
// Block: 256 thr = 8 waves; tile 64 rows x 64 cols; wave = 4 row x 2 col groups,
// 2 WMMAs per wave per k-chunk (A-fragment reuse). Double-buffered LDS with
// async global->LDS prefetch of the next chunk overlapped with WMMA compute.

template <bool FAST>
__global__ void edge_gemm(const float* __restrict__ xin, int ldx, int C,
                          const unsigned short* __restrict__ xbf_in,
                          const int* __restrict__ nnidx,
                          const unsigned short* __restrict__ Wbf, int Kp, int Cout,
                          float* __restrict__ outb,
                          float2* __restrict__ parts, int b) {
  __shared__ __align__(16) unsigned short As[2][64][PITCH];
  __shared__ __align__(16) unsigned short Bs[2][64][PITCH];
  __shared__ float rs[256], rq[256];

  const int tid  = threadIdx.x;
  const int lane = tid & 31;
  const int wave = tid >> 5;
  const int rt = wave & 3;                 // row group (16 rows)
  const int ct = wave >> 2;                // col group (32 cols)
  const int rowBase = blockIdx.x * 64;
  const int colBase = blockIdx.y * 64;

  // builder coords: 64 rows x 4 segments of 8 elems
  const int arow = tid >> 2;
  const int acol0 = (tid & 3) * 8;
  const int grow = rowBase + arow;
  const int n    = grow / KNN;
  const int knb  = grow - n * KNN;
  const int nbr  = nnidx[((size_t)b * NP + n) * KNN + knb];
  const float* xc = xin + ((size_t)b * NP + n)   * ldx;
  const float* xf = xin + ((size_t)b * NP + nbr) * ldx;
  const unsigned short* xbfr =
      FAST ? (xbf_in + ((size_t)b * NP + n) * 512 + acol0) : (const unsigned short*)0;
  const unsigned short* wsrc = Wbf + (size_t)(colBase + arow) * Kp + acol0;

  // issue all loads/stores for chunk t into buffer pb; returns #async per thread
  auto issue_chunk = [&](int t, int pb) -> int {
    const int kk = t * 32;
    async_b128(&Bs[pb][arow][acol0], wsrc + kk);       // B tile (weights, bf16)
    int na = 1;
    if (FAST) {
      if (kk < C) {                                     // diff chunk (uniform)
        const int c0 = kk + acol0;
        float4 f0 = *(const float4*)(xf + c0);
        float4 f1 = *(const float4*)(xf + c0 + 4);
        float4 g0 = *(const float4*)(xc + c0);
        float4 g1 = *(const float4*)(xc + c0 + 4);
        unsigned* dst = (unsigned*)&As[pb][arow][acol0];
        dst[0] = pk2(f0.x - g0.x, f0.y - g0.y);
        dst[1] = pk2(f0.z - g0.z, f0.w - g0.w);
        dst[2] = pk2(f1.x - g1.x, f1.y - g1.y);
        dst[3] = pk2(f1.z - g1.z, f1.w - g1.w);
      } else {                                          // center chunk: async copy
        async_b128(&As[pb][arow][acol0], xbfr + (kk - C));
        na = 2;
      }
    } else {
#pragma unroll
      for (int i = 0; i < 8; ++i) {
        int c = kk + acol0 + i;
        float v = 0.f;
        if (c < C)          v = xf[c] - xc[c];
        else if (c < 2 * C) v = xc[c - C];
        As[pb][arow][acol0 + i] = f2bf(v);
      }
    }
    return na;
  };

  v8f acc0 = {}, acc1 = {};
  const int kIters = Kp >> 5;
  (void)issue_chunk(0, 0);
  for (int t = 0; t < kIters; ++t) {
    const int pb = t & 1;
    int nn = 0;
    if (t + 1 < kIters) nn = issue_chunk(t + 1, pb ^ 1);
    wait_async(nn);                                    // chunk t landed (in-order)
    __syncthreads();                                   // publish chunk t

    const int h  = lane >> 4;
    const int mr = lane & 15;
    v16bf af  = frag_a((const unsigned int*)&As[pb][rt * 16 + mr][0], h);
    v16bf bf0 = frag_b((const unsigned int*)&Bs[pb][ct * 32 + mr][0], h);
    v16bf bf1 = frag_b((const unsigned int*)&Bs[pb][ct * 32 + 16 + mr][0], h);
    acc0 = __builtin_amdgcn_wmma_f32_16x16x32_bf16(false, af, false, bf0,
                                                   (short)0, acc0, false, false);
    acc1 = __builtin_amdgcn_wmma_f32_16x16x32_bf16(false, af, false, bf1,
                                                   (short)0, acc1, false, false);
    __syncthreads();                                   // frags done; buf reusable
  }

  // store + LayerNorm partial stats
  const int h  = lane >> 4;
  const int nc = lane & 15;
  const int oc0 = colBase + ct * 32 + nc;
  const int oc1 = oc0 + 16;
  float s = 0.f, q = 0.f;
#pragma unroll
  for (int r = 0; r < 8; ++r) {
    int rl = rt * 16 + r + 8 * h;
    size_t ro = (size_t)(rowBase + rl) * Cout;
    float v0 = acc0[r], v1 = acc1[r];
    outb[ro + oc0] = v0;
    outb[ro + oc1] = v1;
    s += v0 + v1; q += v0 * v0 + v1 * v1;
  }
  rs[tid] = s; rq[tid] = q;
  __syncthreads();
  for (int st = 128; st > 0; st >>= 1) {
    if (tid < st) { rs[tid] += rs[tid + st]; rq[tid] += rq[tid + st]; }
    __syncthreads();
  }
  if (tid == 0) {
    float2 pr; pr.x = rs[0]; pr.y = rq[0];
    parts[blockIdx.y * gridDim.x + blockIdx.x] = pr;
  }
}

__global__ void reduce_partials(const float2* __restrict__ p, int nblk,
                                float* __restrict__ stat) {
  __shared__ float rs[256], rq[256];
  float s = 0.f, q = 0.f;
  for (int i = threadIdx.x; i < nblk; i += 256) { s += p[i].x; q += p[i].y; }
  rs[threadIdx.x] = s; rq[threadIdx.x] = q;
  __syncthreads();
  for (int st = 128; st > 0; st >>= 1) {
    if (threadIdx.x < st) {
      rs[threadIdx.x] += rs[threadIdx.x + st];
      rq[threadIdx.x] += rq[threadIdx.x + st];
    }
    __syncthreads();
  }
  if (threadIdx.x == 0) { stat[0] = rs[0]; stat[1] = rq[0]; }
}

// ---------------- LayerNorm + affine + LeakyReLU + max over k --------------

__global__ void ln_apply(const float* __restrict__ outb,
                         const float* __restrict__ lnw,
                         const float* __restrict__ lnb,
                         const float* __restrict__ stat, float invcount,
                         int Cout, float* __restrict__ xdst,
                         unsigned short* __restrict__ xbf_dst, int b) {
  int t = blockIdx.x * 256 + threadIdx.x;          // N*Cout threads
  int n = t / Cout;
  int o = t - n * Cout;
  float mu   = stat[0] * invcount;
  float var  = stat[1] * invcount - mu * mu;
  float rstd = rsqrtf(var + 1e-5f);
  float mx = -3.402823466e38f;
  for (int k = 0; k < KNN; ++k) {
    float v = outb[(size_t)(n * KNN + k) * Cout + o];
    v = (v - mu) * rstd;
    size_t li = ((size_t)o * NP + n) * KNN + k;
    v = v * lnw[li] + lnb[li];
    v = v >= 0.f ? v : 0.2f * v;
    mx = fmaxf(mx, v);
  }
  size_t di = ((size_t)b * NP + n) * 512 + o;
  xdst[di]    = mx;
  xbf_dst[di] = f2bf(mx);
}

// ---------------- final linear: fully async A & B, double-buffered ---------

__global__ void gemm_x5(const unsigned short* __restrict__ xbf,
                        const unsigned short* __restrict__ w5bf,
                        float* __restrict__ x5) {
  __shared__ __align__(16) unsigned short As[2][64][PITCH];
  __shared__ __align__(16) unsigned short Bs[2][64][PITCH];

  const int tid  = threadIdx.x;
  const int lane = tid & 31;
  const int wave = tid >> 5;
  const int rt = wave & 3;
  const int ct = wave >> 2;
  const int rowBase = blockIdx.x * 64;
  const int colBase = blockIdx.y * 64;

  const int arow = tid >> 2;
  const int acol0 = (tid & 3) * 8;
  const unsigned short* ar = xbf  + (size_t)(rowBase + arow) * 512 + acol0;
  const unsigned short* br = w5bf + (size_t)(colBase + arow) * 512 + acol0;

  auto issue_chunk = [&](int t, int pb) {
    const int kk = t * 32;
    async_b128(&As[pb][arow][acol0], ar + kk);
    async_b128(&Bs[pb][arow][acol0], br + kk);
  };

  v8f acc0 = {}, acc1 = {};
  issue_chunk(0, 0);
  for (int t = 0; t < 16; ++t) {                   // 512 / 32
    const int pb = t & 1;
    int nn = 0;
    if (t + 1 < 16) { issue_chunk(t + 1, pb ^ 1); nn = 2; }
    wait_async(nn);
    __syncthreads();
    const int h  = lane >> 4;
    const int mr = lane & 15;
    v16bf af  = frag_a((const unsigned int*)&As[pb][rt * 16 + mr][0], h);
    v16bf bf0 = frag_b((const unsigned int*)&Bs[pb][ct * 32 + mr][0], h);
    v16bf bf1 = frag_b((const unsigned int*)&Bs[pb][ct * 32 + 16 + mr][0], h);
    acc0 = __builtin_amdgcn_wmma_f32_16x16x32_bf16(false, af, false, bf0,
                                                   (short)0, acc0, false, false);
    acc1 = __builtin_amdgcn_wmma_f32_16x16x32_bf16(false, af, false, bf1,
                                                   (short)0, acc1, false, false);
    __syncthreads();
  }
  const int h  = lane >> 4;
  const int nc = lane & 15;
  const int oc0 = colBase + ct * 32 + nc;
#pragma unroll
  for (int r = 0; r < 8; ++r) {
    int rl = rt * 16 + r + 8 * h;
    size_t ro = (size_t)(rowBase + rl) * 1024;
    x5[ro + oc0]      = acc0[r];
    x5[ro + oc0 + 16] = acc1[r];
  }
}

// ---------------- BatchNorm stats (per channel over B*N) -------------------

__global__ void bn_stats(const float* __restrict__ x5, float* __restrict__ bnst) {
  __shared__ float rs[256], rq[256];
  const int o = blockIdx.x;
  float s = 0.f, q = 0.f;
  for (int r = threadIdx.x; r < NB * NP; r += 256) {
    float v = x5[(size_t)r * 1024 + o];
    s += v; q += v * v;
  }
  rs[threadIdx.x] = s; rq[threadIdx.x] = q;
  __syncthreads();
  for (int st = 128; st > 0; st >>= 1) {
    if (threadIdx.x < st) {
      rs[threadIdx.x] += rs[threadIdx.x + st];
      rq[threadIdx.x] += rq[threadIdx.x + st];
    }
    __syncthreads();
  }
  if (threadIdx.x == 0) {
    float mu  = rs[0] * (1.f / (NB * NP));
    float var = rq[0] * (1.f / (NB * NP)) - mu * mu;
    bnst[o]        = mu;
    bnst[1024 + o] = rsqrtf(var + 1e-5f);
  }
}

// ---------------- BN + LeakyReLU + max/mean pooling ------------------------

__global__ void finalize(const float* __restrict__ x5,
                         const float* __restrict__ bnst,
                         const float* __restrict__ bw,
                         const float* __restrict__ bb,
                         float* __restrict__ out) {
  int t = blockIdx.x * 256 + threadIdx.x;          // B*1024 threads
  int b = t >> 10;
  int o = t & 1023;
  float mu = bnst[o], rstd = bnst[1024 + o], w = bw[o], bi = bb[o];
  float mx = -3.402823466e38f, sm = 0.f;
  for (int n = 0; n < NP; ++n) {
    float v = x5[((size_t)b * NP + n) * 1024 + o];
    v = (v - mu) * rstd * w + bi;
    v = v >= 0.f ? v : 0.2f * v;
    mx = fmaxf(mx, v);
    sm += v;
  }
  out[b * 2048 + o]        = mx;
  out[b * 2048 + 1024 + o] = sm * (1.f / NP);
}

// ---------------- host driver ----------------------------------------------

extern "C" void kernel_launch(void* const* d_in, const int* in_sizes, int n_in,
                              void* d_out, int out_size, void* d_ws, size_t ws_size,
                              hipStream_t stream) {
  (void)in_sizes; (void)n_in; (void)out_size; (void)ws_size;

  const float* x    = (const float*)d_in[0];
  const float* W1   = (const float*)d_in[1];
  const float* ln1w = (const float*)d_in[2];
  const float* ln1b = (const float*)d_in[3];
  const float* W2   = (const float*)d_in[4];
  const float* ln2w = (const float*)d_in[5];
  const float* ln2b = (const float*)d_in[6];
  const float* W3   = (const float*)d_in[7];
  const float* ln3w = (const float*)d_in[8];
  const float* ln3b = (const float*)d_in[9];
  const float* W4   = (const float*)d_in[10];
  const float* ln4w = (const float*)d_in[11];
  const float* ln4b = (const float*)d_in[12];
  const float* W5   = (const float*)d_in[13];
  const float* bn5w = (const float*)d_in[14];
  const float* bn5b = (const float*)d_in[15];

  char* ws = (char*)d_ws;
  // workspace layout (bytes, all 256B-aligned)
  constexpr size_t XFEAT_OFF = 0;                                   // fp32 feats (B,N,512)
  constexpr size_t XBF_OFF   = XFEAT_OFF + (size_t)NB*NP*512*4;     // bf16 mirror
  constexpr size_t IDX_OFF   = XBF_OFF   + (size_t)NB*NP*512*2;     // knn idx
  constexpr size_t OUTB_OFF  = IDX_OFF   + (size_t)NB*NP*KNN*4;     // per-batch pre-LN
  constexpr size_t X5_OFF    = OUTB_OFF  + (size_t)NP*KNN*256*4;    // (B,N,1024)
  constexpr size_t NORM_OFF  = X5_OFF    + (size_t)NB*NP*1024*4;
  constexpr size_t STAT_OFF  = NORM_OFF  + (size_t)NB*NP*4;
  constexpr size_t PART_OFF  = STAT_OFF  + 256;
  constexpr size_t BNST_OFF  = PART_OFF  + 16384;
  constexpr size_t WBF_OFF   = BNST_OFF  + 8192;                    // stage weights bf16
  constexpr size_t W5BF_OFF  = WBF_OFF   + 131072;                  // 1024*512 bf16

  float*          xfeat = (float*)(ws + XFEAT_OFF);
  unsigned short* xbf   = (unsigned short*)(ws + XBF_OFF);
  int*            nnidx = (int*)(ws + IDX_OFF);
  float*          outb  = (float*)(ws + OUTB_OFF);
  float*          x5    = (float*)(ws + X5_OFF);
  float*          norms = (float*)(ws + NORM_OFF);
  float*          stat  = (float*)(ws + STAT_OFF);
  float2*         parts = (float2*)(ws + PART_OFF);
  float*          bnst  = (float*)(ws + BNST_OFF);
  unsigned short* wbf   = (unsigned short*)(ws + WBF_OFF);
  unsigned short* w5bf  = (unsigned short*)(ws + W5BF_OFF);

  auto stage = [&](const float* xin, int ldx, int C, int inOff, const float* W,
                   const float* lnw, const float* lnb, int Cout, int outOff) {
    if (C == 3) {
      norms_kernel<3><<<64, 256, 0, stream>>>(xin, ldx, norms);
      knn_kernel<3><<<128, 128, 0, stream>>>(xin, ldx, norms, nnidx);
    } else if (C == 64) {
      norms_kernel<64><<<64, 256, 0, stream>>>(xin, ldx, norms);
      knn_kernel<64><<<128, 128, 0, stream>>>(xin, ldx, norms, nnidx);
    } else {
      norms_kernel<128><<<64, 256, 0, stream>>>(xin, ldx, norms);
      knn_kernel<128><<<128, 128, 0, stream>>>(xin, ldx, norms, nnidx);
    }
    const int Cin = 2 * C;
    const int Kp  = ((Cin + 31) / 32) * 32;
    cvt_w<<<(Cout * Kp + 255) / 256, 256, 0, stream>>>(W, Cin, Kp, wbf, Cout * Kp);
    const int gx = NP * KNN / 64;                   // 320
    const int gy = Cout / 64;
    const float invcount = 1.f / ((float)Cout * NP * KNN);
    for (int b = 0; b < NB; ++b) {
      if (C == 3)
        edge_gemm<false><<<dim3(gx, gy), 256, 0, stream>>>(
            xin, ldx, C, xbf, nnidx, wbf, Kp, Cout, outb, parts, b);
      else
        edge_gemm<true><<<dim3(gx, gy), 256, 0, stream>>>(
            xin, ldx, C, xbf + inOff, nnidx, wbf, Kp, Cout, outb, parts, b);
      reduce_partials<<<1, 256, 0, stream>>>(parts, gx * gy, stat + 2 * b);
      ln_apply<<<(NP * Cout) / 256, 256, 0, stream>>>(
          outb, lnw, lnb, stat + 2 * b, invcount, Cout,
          xfeat + outOff, xbf + outOff, b);
    }
  };

  stage(x,           3,   3,   0,   W1, ln1w, ln1b, 64,  0);    // -> cols 0..63
  stage(xfeat,       512, 64,  0,   W2, ln2w, ln2b, 64,  64);   // -> cols 64..127
  stage(xfeat + 64,  512, 64,  64,  W3, ln3w, ln3b, 128, 128);  // -> cols 128..255
  stage(xfeat + 128, 512, 128, 128, W4, ln4w, ln4b, 256, 256);  // -> cols 256..511

  cvt_w<<<(1024 * 512 + 255) / 256, 256, 0, stream>>>(W5, 512, 512, w5bf, 1024 * 512);
  gemm_x5<<<dim3(NB * NP / 64, 16), 256, 0, stream>>>(xbf, w5bf, x5);
  bn_stats<<<1024, 256, 0, stream>>>(x5, bnst);
  finalize<<<NB * 1024 / 256, 256, 0, stream>>>(x5, bnst, bn5w, bn5b,
                                                (float*)d_out);
}
